// VposeFiled_Vjmlp_52467320488329
// MI455X (gfx1250) — matrix operand order repaired and checked by reading, not compiled, not run on hardware
//
#include <hip/hip_runtime.h>
#include <hip/hip_bf16.h>
#include <math.h>

// ---------------------------------------------------------------------------
// Problem constants (from reference)
// ---------------------------------------------------------------------------
#define VJ    80
#define FS    480
#define HD    256
#define RFFK  240
#define BATCH 2048
#define TWO_PI 6.283185307179586f
#define SLOPE  0.01f

typedef __attribute__((ext_vector_type(16))) _Float16 v16h;
typedef __attribute__((ext_vector_type(8)))  _Float16 v8h;
typedef __attribute__((ext_vector_type(8)))  float    v8f;

// ---------------------------------------------------------------------------
// Workspace layout (bytes). Total ~428 MB.
//   f16 weight-normed weights for layers 0..2 (fit in 192MB L2 -> batch reuse),
//   f32 weights for tiny layer 3, two ping-pong f16 activation buffers.
// ---------------------------------------------------------------------------
#define OFF_WN0  ((size_t)0)                 // 80*512*480*2 = 39,321,600
#define OFF_WN1  ((size_t)39321600)          // 80*512*512*2 = 41,943,040
#define OFF_WN2  ((size_t)81264640)          // 80*256*512*2 = 20,971,520
#define OFF_WN3  ((size_t)102236160)         // 80*6*256*4   =    491,520
#define OFF_ACT0 ((size_t)102727680)         // 80*2048*480*2 = 157,286,400 (x0, later a2)
#define OFF_ACT1 ((size_t)260014080)         // 80*2048*512*2 = 167,772,160 (a1, later a3)

// ---------------------------------------------------------------------------
// Kernel 1: weight_norm fold + precision convert.
//   w[o,:] = v[o,:] * g[o] / ||v[o,:]||   -> out (f16 for WMMA layers, f32 L3)
// One block per output row; block reduction for the row norm.
// ---------------------------------------------------------------------------
template <typename T>
__global__ void prep_weights(const float* __restrict__ v,
                             const float* __restrict__ gw,
                             T* __restrict__ out, int K)
{
    __shared__ float red[256];
    const int row = blockIdx.x;
    const int tid = threadIdx.x;
    const float* vr = v + (size_t)row * K;

    float s = 0.f;
    for (int i = tid; i < K; i += 256) { float x = vr[i]; s += x * x; }
    red[tid] = s;
    __syncthreads();
    for (int off = 128; off > 0; off >>= 1) {
        if (tid < off) red[tid] += red[tid + off];
        __syncthreads();
    }
    const float scale = gw[row] * rsqrtf(red[0]);

    T* orow = out + (size_t)row * K;
    for (int i = tid; i < K; i += 256) orow[i] = (T)(vr[i] * scale);
}

// ---------------------------------------------------------------------------
// Kernel 2: RFF encode + add point_features + transpose(-2,-1).reshape shuffle.
// Output x0 in GEMM layout: x0[((g*BATCH)+b)*FS + i]  (f16)
// Mapping: flat = g*FS+i ; k = flat/VJ ; n = flat%VJ ; value = pf[b,n,k]+feat
//   feat = cos(emb[b,n,k])        (k <  240)
//        = sin(emb[b,n,k-240])    (k >= 240)
//   emb[b,n,j] = 2*pi * sum_d qp[b,n,d] * Brff[d,j]
// ---------------------------------------------------------------------------
__global__ void build_x0(const float* __restrict__ qp,
                         const float* __restrict__ pf,
                         const float* __restrict__ Brff,
                         _Float16* __restrict__ x0)
{
    const size_t total = (size_t)VJ * BATCH * FS;
    size_t idx = (size_t)blockIdx.x * 256 + threadIdx.x;
    if (idx >= total) return;

    int i = (int)(idx % FS);
    size_t t = idx / FS;
    int b = (int)(t % BATCH);
    int g = (int)(t / BATCH);

    int flat = g * FS + i;           // position in transposed-flat space
    int k = flat / VJ;               // original feature index 0..479
    int n = flat % VJ;               // original group index  0..79
    int j = (k < RFFK) ? k : (k - RFFK);

    const float* q = qp + ((size_t)b * VJ + n) * 3;
    float emb = TWO_PI * (q[0] * Brff[j] +
                          q[1] * Brff[RFFK + j] +
                          q[2] * Brff[2 * RFFK + j]);
    float feat = (k < RFFK) ? cosf(emb) : sinf(emb);
    float val  = pf[((size_t)b * VJ + n) * FS + k] + feat;
    x0[idx] = (_Float16)val;
}

// ---------------------------------------------------------------------------
// Kernel 3: grouped GEMM + bias + LeakyReLU, f16 in / f32 acc / f16 out.
//   For group g:  Out[b, o] = lrelu( sum_i A[b,i] * W[o,i] + bias[o] )
//   A : [g][BATCH][K] row-major,  W : [g][N][K] row-major (K contiguous = WMMA
//   B-fragment friendly),  Out : [g][BATCH][N].
// Block: 256 thr (8 waves), tile M=128 x N=128; wave tile 32x64;
// K staged in LDS 32 at a time; 8 x v_wmma_f32_16x16x32_f16 per K-step.
// ---------------------------------------------------------------------------
template <int K, int N>
__global__ __launch_bounds__(256)
void gemm_grouped_lrelu(const _Float16* __restrict__ A,
                        const _Float16* __restrict__ W,
                        const float* __restrict__ bias,
                        _Float16* __restrict__ Out)
{
    // 40-halfword row pitch (pad 8) to stagger LDS banks; rows stay 16B aligned.
    __shared__ _Float16 As[128 * 40];
    __shared__ _Float16 Bs[128 * 40];

    const int g   = blockIdx.z;
    const int bm0 = blockIdx.y * 128;      // batch-tile origin
    const int nb0 = blockIdx.x * 128;      // out-feature-tile origin
    const int tid = threadIdx.x;
    const int lane  = tid & 31;
    const int wv    = tid >> 5;            // wave 0..7
    const int wm    = wv & 3;              // 4 waves along M (32 rows each)
    const int wn    = wv >> 2;             // 2 waves along N (64 cols each)
    const int lrow  = lane & 15;
    const int khalf = lane >> 4;

    const _Float16* Ag = A + (size_t)g * BATCH * K;
    const _Float16* Wg = W + (size_t)g * N * K;

    const v8f vzero = {};
    v8f acc[2][4];
#pragma unroll
    for (int mi = 0; mi < 2; ++mi)
#pragma unroll
        for (int ni = 0; ni < 4; ++ni) acc[mi][ni] = vzero;

    const int KT = K / 32;
    for (int kt = 0; kt < KT; ++kt) {
        const int k0 = kt * 32;
        __syncthreads();
        // Cooperative stage of A(128x32) and B(128x32) tiles: b128 per chunk.
#pragma unroll
        for (int it = 0; it < 2; ++it) {
            int t = tid + it * 256;        // 0..511
            int r = t >> 2;                // 0..127
            int c = (t & 3) << 3;          // 0,8,16,24
            *(v8h*)&As[r * 40 + c] =
                *(const v8h*)&Ag[(size_t)(bm0 + r) * K + (k0 + c)];
            *(v8h*)&Bs[r * 40 + c] =
                *(const v8h*)&Wg[(size_t)(nb0 + r) * K + (k0 + c)];
        }
        __syncthreads();

        // Prefetch next K-tile toward L2/L0 (global_prefetch_b8 on gfx1250).
        if (kt + 1 < KT) {
            int r = tid >> 2;
            int c = (tid & 3) << 3;
            __builtin_prefetch(&Ag[(size_t)(bm0 + r) * K + (k0 + 32 + c)], 0, 3);
            __builtin_prefetch(&Wg[(size_t)(nb0 + r) * K + (k0 + 32 + c)], 0, 3);
        }

        // A fragments (ISA 16-bit A 16x32 layout): lane(row=lrow, khalf)
        // holds K = [8*khalf, +8) in v0..3 and K = [16+8*khalf, +8) in v4..7.
        v16h afrag[2];
#pragma unroll
        for (int mi = 0; mi < 2; ++mi) {
            int ar = wm * 32 + mi * 16 + lrow;
            v8h lo = *(const v8h*)&As[ar * 40 + khalf * 8];
            v8h hi = *(const v8h*)&As[ar * 40 + 16 + khalf * 8];
            afrag[mi] = __builtin_shufflevector(lo, hi,
                0, 1, 2, 3, 4, 5, 6, 7, 8, 9, 10, 11, 12, 13, 14, 15);
        }
        // B fragments (32x16): lane(col=lrow, khalf) holds K = [8*khalf, +8)
        // in v0..3 and K = [16+8*khalf, +8) in v4..7 (same order as A).
        v16h bfrag[4];
#pragma unroll
        for (int ni = 0; ni < 4; ++ni) {
            int bc = wn * 64 + ni * 16 + lrow;
            v8h lo = *(const v8h*)&Bs[bc * 40 + khalf * 8];
            v8h hi = *(const v8h*)&Bs[bc * 40 + 16 + khalf * 8];
            bfrag[ni] = __builtin_shufflevector(lo, hi,
                0, 1, 2, 3, 4, 5, 6, 7, 8, 9, 10, 11, 12, 13, 14, 15);
        }

#pragma unroll
        for (int mi = 0; mi < 2; ++mi)
#pragma unroll
            for (int ni = 0; ni < 4; ++ni)
                acc[mi][ni] = __builtin_amdgcn_wmma_f32_16x16x32_f16(
                    false, afrag[mi], false, bfrag[ni],
                    (short)0, acc[mi][ni], false, false);
    }

    // Epilogue: bias + LeakyReLU, f16 store.
    // C/D layout: VGPR j, lanes0-15 -> M=j, lanes16-31 -> M=8+j, N=lrow.
#pragma unroll
    for (int ni = 0; ni < 4; ++ni) {
        int col = nb0 + wn * 64 + ni * 16 + lrow;
        float bv = bias[g * N + col];
#pragma unroll
        for (int mi = 0; mi < 2; ++mi) {
#pragma unroll
            for (int j = 0; j < 8; ++j) {
                int row = bm0 + wm * 32 + mi * 16 + khalf * 8 + j;
                float val = acc[mi][ni][j] + bv;
                val = (val >= 0.f) ? val : SLOPE * val;
                Out[((size_t)g * BATCH + row) * N + col] = (_Float16)val;
            }
        }
    }
}

// ---------------------------------------------------------------------------
// Kernel 4: tiny final layer 256 -> 6, fp32, no activation, to d_out.
//   out[(b*VJ+g)*6+o] = b3[g*6+o] + sum_i a3[g][b][i] * w3[(g*6+o)*256+i]
// ---------------------------------------------------------------------------
__global__ void final_layer(const _Float16* __restrict__ a3,
                            const float* __restrict__ w3,
                            const float* __restrict__ b3,
                            float* __restrict__ out)
{
    const size_t total = (size_t)BATCH * VJ * 6;
    size_t idx = (size_t)blockIdx.x * 256 + threadIdx.x;
    if (idx >= total) return;

    int o = (int)(idx % 6);
    size_t t = idx / 6;
    int g = (int)(t % VJ);
    int b = (int)(t / VJ);

    const _Float16* ar = a3 + ((size_t)g * BATCH + b) * HD;
    const float*    wr = w3 + ((size_t)g * 6 + o) * HD;
    float s = b3[g * 6 + o];
#pragma unroll 8
    for (int i = 0; i < HD; ++i) s += (float)ar[i] * wr[i];
    out[idx] = s;
}

// ---------------------------------------------------------------------------
// Host launcher
// ---------------------------------------------------------------------------
extern "C" void kernel_launch(void* const* d_in, const int* in_sizes, int n_in,
                              void* d_out, int out_size, void* d_ws, size_t ws_size,
                              hipStream_t stream)
{
    (void)in_sizes; (void)n_in; (void)out_size; (void)ws_size;

    const float* qp   = (const float*)d_in[0];
    const float* pf   = (const float*)d_in[1];
    const float* Brff = (const float*)d_in[2];
    const float* v0 = (const float*)d_in[3];
    const float* g0 = (const float*)d_in[4];
    const float* b0 = (const float*)d_in[5];
    const float* v1 = (const float*)d_in[6];
    const float* g1 = (const float*)d_in[7];
    const float* b1 = (const float*)d_in[8];
    const float* v2 = (const float*)d_in[9];
    const float* g2 = (const float*)d_in[10];
    const float* b2 = (const float*)d_in[11];
    const float* v3 = (const float*)d_in[12];
    const float* g3 = (const float*)d_in[13];
    const float* b3 = (const float*)d_in[14];

    char* ws = (char*)d_ws;
    _Float16* wn0  = (_Float16*)(ws + OFF_WN0);
    _Float16* wn1  = (_Float16*)(ws + OFF_WN1);
    _Float16* wn2  = (_Float16*)(ws + OFF_WN2);
    float*    wn3  = (float*)   (ws + OFF_WN3);
    _Float16* act0 = (_Float16*)(ws + OFF_ACT0);   // x0, then a2
    _Float16* act1 = (_Float16*)(ws + OFF_ACT1);   // a1, then a3

    // 1) weight-norm fold + convert
    prep_weights<_Float16><<<VJ * 512, 256, 0, stream>>>(v0, g0, wn0, FS);
    prep_weights<_Float16><<<VJ * 512, 256, 0, stream>>>(v1, g1, wn1, 2 * HD);
    prep_weights<_Float16><<<VJ * 256, 256, 0, stream>>>(v2, g2, wn2, 2 * HD);
    prep_weights<float>   <<<VJ * 6,   256, 0, stream>>>(v3, g3, wn3, HD);

    // 2) RFF + shuffle -> x0 (f16)
    {
        size_t total = (size_t)VJ * BATCH * FS;
        int blocks = (int)((total + 255) / 256);
        build_x0<<<blocks, 256, 0, stream>>>(qp, pf, Brff, act0);
    }

    // 3) three WMMA grouped-GEMM layers (ping-pong activations)
    gemm_grouped_lrelu<FS,     512><<<dim3(4, BATCH / 128, VJ), 256, 0, stream>>>(
        act0, wn0, b0, act1);                       // x0 -> a1
    gemm_grouped_lrelu<2 * HD, 512><<<dim3(4, BATCH / 128, VJ), 256, 0, stream>>>(
        act1, wn1, b1, act0);                       // a1 -> a2
    gemm_grouped_lrelu<2 * HD, 256><<<dim3(2, BATCH / 128, VJ), 256, 0, stream>>>(
        act0, wn2, b2, act1);                       // a2 -> a3

    // 4) final 256->6 fp32 layer to d_out [B, VJ, 6]
    {
        size_t total = (size_t)BATCH * VJ * 6;
        int blocks = (int)((total + 255) / 256);
        final_layer<<<blocks, 256, 0, stream>>>(act1, wn3, b3, (float*)d_out);
    }
}
